// TwinAttentionMatchingLoss_60876866454280
// MI455X (gfx1250) — matrix operand order repaired and checked by reading, not compiled, not run on hardware
//
#include <hip/hip_runtime.h>
#include <stdint.h>

typedef __attribute__((ext_vector_type(16))) __bf16 v16bf;
typedef __attribute__((ext_vector_type(8)))  float  v8f;

#define BB    16
#define NN1   1024
#define NN2   1024
#define DD    256
#define WPB   8                 // waves per block (256 threads = 1 WGP-filling WG)
#define TILES (NN2 / 16)        // 64 column tiles
#define LSTR  264               // padded LDS stride per column (bf16 elems): 264*2B = 33*16B

// sched_group_barrier masks (AMDGPU IGroupLP)
#define SG_MFMA    0x008        // MFMA/WMMA
#define SG_DSREAD  0x100        // DS read
#if __has_builtin(__builtin_amdgcn_sched_group_barrier)
#define SGB(mask, size, id) __builtin_amdgcn_sched_group_barrier((mask), (size), (id))
#else
#define SGB(mask, size, id)
#endif

// ---- fp32 -> bf16 hi/lo split (truncation; lo captures the residual) ----
static __device__ __forceinline__ __bf16 bf_trunc(float x, float& hif) {
  union { float f; uint32_t u; } c; c.f = x;
  union { float f; uint32_t u; } h; h.u = c.u & 0xffff0000u;
  hif = h.f;
  union { uint16_t s; __bf16 b; } r; r.s = (uint16_t)(c.u >> 16);
  return r.b;
}

template <int BASE>
static __device__ __forceinline__ void split4(const float4 f, v16bf& hi, v16bf& lo, float scale) {
  const float xs[4] = { f.x * scale, f.y * scale, f.z * scale, f.w * scale };
#pragma unroll
  for (int j = 0; j < 4; ++j) {
    float hf;
    hi[BASE + j] = bf_trunc(xs[j], hf);
    float dummy;
    lo[BASE + j] = bf_trunc(xs[j] - hf, dummy);
  }
}

__global__ void twin_loss_init(float* __restrict__ accum) {
  accum[0] = 0.0f;  // loss numerator
  accum[1] = 0.0f;  // mask count
}

__global__ __launch_bounds__(32 * WPB)
void twin_loss_main(const float* __restrict__ z1m,
                    const float* __restrict__ z2m,
                    const int*   __restrict__ midx,
                    const float* __restrict__ temp,
                    const uint8_t* __restrict__ mask1,
                    float* __restrict__ accum)
{
  // double-buffered B tile in bf16 hi/lo, [col][k] with padded stride
  __shared__ __bf16 sHi[2][16 * LSTR];
  __shared__ __bf16 sLo[2][16 * LSTR];

  const int b    = blockIdx.x >> 3;            // batch
  const int g    = blockIdx.x & 7;             // row-tile group (8 tiles per block)
  const int tid  = (int)threadIdx.x;
  const int w    = tid >> 5;                   // wave id 0..7
  const int lane = tid & 31;
  const int half = lane >> 4;                  // 0: lanes 0-15, 1: lanes 16-31
  const int l    = lane & 15;
  const int m0   = (g * WPB + w) << 4;         // this wave's 16-row tile base

  const float invT = 1.0f / temp[0];           // fold temperature into A

  // ---- stage this wave's A tile (16 rows x 256 K) in registers, WMMA A layout ----
  // lane<16 : elems 0-15 -> K=kc+0..15
  // lane>=16: elems 0-15 -> K=kc+16..31
  const float* arow = z1m + ((size_t)b * NN1 + (size_t)(m0 + l)) * DD;
  v16bf a_hi[8], a_lo[8];
#pragma unroll
  for (int c = 0; c < 8; ++c) {
    const int kb = c * 32 + half * 16;
    const float4 f0 = *(const float4*)(arow + kb +  0);
    const float4 f1 = *(const float4*)(arow + kb +  4);
    const float4 f2 = *(const float4*)(arow + kb +  8);
    const float4 f3 = *(const float4*)(arow + kb + 12);
    split4<0 >(f0, a_hi[c], a_lo[c], invT);
    split4<4 >(f1, a_hi[c], a_lo[c], invT);
    split4<8 >(f2, a_hi[c], a_lo[c], invT);
    split4<12>(f3, a_hi[c], a_lo[c], invT);
  }

  // ---- per-row metadata + per-LANE streaming logsumexp state ----
  int   cidx[8];
  float wgt[8], mskf[8], mrun[8], srun[8], gacc[8];
#pragma unroll
  for (int v = 0; v < 8; ++v) {
    const int  row = m0 + half * 8 + v;        // C/D layout: half0 -> rows v, half1 -> rows v+8
    const int  raw = midx[b * NN1 + row];
    const bool ism = raw < (NN2 - 1);
    cidx[v] = ism ? raw : (NN2 - 1);
    mskf[v] = mask1[b * NN1 + row] ? 1.0f : 0.0f;
    wgt[v]  = (ism ? 1.0f : 0.5f) * mskf[v];
    mrun[v] = -__builtin_inff();
    srun[v] = 0.0f;
    gacc[v] = 0.0f;
  }

  // ---- cooperative fill roles: thread -> (column, K-sixteenth) of the B tile ----
  const int fcol = tid >> 4;                   // 0..15
  const int fks  = (tid & 15) << 4;            // 0,16,...,240
  const float* fsrc = z2m + ((size_t)b * NN2 + (size_t)fcol) * DD + fks;

  // prologue: fill tile 0 into buffer 0
  {
    const float4 f0 = *(const float4*)(fsrc +  0);
    const float4 f1 = *(const float4*)(fsrc +  4);
    const float4 f2 = *(const float4*)(fsrc +  8);
    const float4 f3 = *(const float4*)(fsrc + 12);
    v16bf hi, lo;
    split4<0 >(f0, hi, lo, 1.0f);
    split4<4 >(f1, hi, lo, 1.0f);
    split4<8 >(f2, hi, lo, 1.0f);
    split4<12>(f3, hi, lo, 1.0f);
    *(v16bf*)&sHi[0][fcol * LSTR + fks] = hi;
    *(v16bf*)&sLo[0][fcol * LSTR + fks] = lo;
  }
  __syncthreads();

  for (int t = 0; t < TILES; ++t) {
    const int  buf       = t & 1;
    const bool have_next = (t + 1) < TILES;

    // issue next tile's global loads early (latency hides behind the WMMA chain)
    float4 nf0, nf1, nf2, nf3;
    if (have_next) {
      const float* s = fsrc + (size_t)(t + 1) * (16 * DD);
      nf0 = *(const float4*)(s +  0);
      nf1 = *(const float4*)(s +  4);
      nf2 = *(const float4*)(s +  8);
      nf3 = *(const float4*)(s + 12);
      if (t + 2 < TILES) __builtin_prefetch(fsrc + (size_t)(t + 2) * (16 * DD), 0, 1);
    }

    // ---- 16x16 score tile: 8 K-chunks x (hi*hi + lo*hi + hi*lo) ----
    // B layout used here (contiguous k, same as A): lane<16 -> col N=l, K=kc+0..15 ; lane>=16 -> col N=l, K=kc+16..31
    const __bf16* bh_base = &sHi[buf][l * LSTR + half * 16];
    const __bf16* bl_base = &sLo[buf][l * LSTR + half * 16];
    v8f acc = {};
    {
      v16bf bh[2], bl[2];
      bh[0] = *(const v16bf*)(bh_base + 0);
      bl[0] = *(const v16bf*)(bl_base + 0);
#pragma unroll
      for (int c = 0; c < 8; ++c) {
        const int cur = c & 1;
        const int nxt = cur ^ 1;
        if (c + 1 < 8) {
          bh[nxt] = *(const v16bf*)(bh_base + (c + 1) * 32);
          bl[nxt] = *(const v16bf*)(bl_base + (c + 1) * 32);
        }
        acc = __builtin_amdgcn_wmma_f32_16x16x32_bf16(false, a_hi[c], false, bh[cur], (short)0, acc, false, false);
        acc = __builtin_amdgcn_wmma_f32_16x16x32_bf16(false, a_lo[c], false, bh[cur], (short)0, acc, false, false);
        acc = __builtin_amdgcn_wmma_f32_16x16x32_bf16(false, a_hi[c], false, bl[cur], (short)0, acc, false, false);
      }
    }
    // Pipeline the unrolled chunk loop: keep DS reads one chunk ahead of the WMMAs
    // (4 ds_load_b128 per chunk, 3 WMMAs per chunk). Forces distinct operand regs
    // and turns per-chunk `s_wait_dscnt 0x0` into partial waits.
    SGB(SG_DSREAD, 4, 0);                      // chunk 0 loads
#pragma unroll
    for (int c = 0; c < 7; ++c) {
      SGB(SG_DSREAD, 4, 0);                    // chunk c+1 loads ahead
      SGB(SG_MFMA,   3, 0);                    // chunk c WMMAs
    }
    SGB(SG_MFMA, 3, 0);                        // chunk 7 WMMAs

    // ---- per-lane online logsumexp + match gather (no cross-lane ops here) ----
#pragma unroll
    for (int v = 0; v < 8; ++v) {
      const float x  = acc[v];                         // already scaled by 1/T via A
      const float mn = fmaxf(mrun[v], x);
      srun[v] = srun[v] * __expf(mrun[v] - mn) + __expf(x - mn);
      mrun[v] = mn;
      gacc[v] += (((t << 4) + l) == cidx[v]) ? x : 0.0f;
    }

    // ---- split + store next tile into the other buffer ----
    if (have_next) {
      v16bf hi, lo;
      split4<0 >(nf0, hi, lo, 1.0f);
      split4<4 >(nf1, hi, lo, 1.0f);
      split4<8 >(nf2, hi, lo, 1.0f);
      split4<12>(nf3, hi, lo, 1.0f);
      *(v16bf*)&sHi[buf ^ 1][fcol * LSTR + fks] = hi;
      *(v16bf*)&sLo[buf ^ 1][fcol * LSTR + fks] = lo;
    }
    __syncthreads();
  }

  // ---- once: merge per-lane (m,s) pairs and gather sums across each 16-lane half ----
  float lossloc = 0.0f, cntloc = 0.0f;
#pragma unroll
  for (int v = 0; v < 8; ++v) {
    float m = mrun[v], s = srun[v], gsum = gacc[v];
#pragma unroll
    for (int off = 1; off < 16; off <<= 1) {
      const float m2 = __shfl_xor(m, off, 32);
      const float s2 = __shfl_xor(s, off, 32);
      gsum += __shfl_xor(gsum, off, 32);
      const float mn = fmaxf(m, m2);
      s = s * __expf(m - mn) + s2 * __expf(m2 - mn);
      m = mn;
    }
    const float lse = m + __logf(s);
    lossloc += (lse - gsum) * wgt[v];          // = -(log_prob) * weight
    cntloc  += mskf[v];
  }
  if (l == 0) {                                // lanes 0 and 16 (one per half)
    atomicAdd(&accum[0], lossloc);
    atomicAdd(&accum[1], cntloc);
  }
}

__global__ void twin_loss_fini(const float* __restrict__ accum, float* __restrict__ out) {
  const float cnt = accum[1];
  out[0] = (cnt > 0.0f) ? (accum[0] / cnt) : 0.0f;
}

extern "C" void kernel_launch(void* const* d_in, const int* in_sizes, int n_in,
                              void* d_out, int out_size, void* d_ws, size_t ws_size,
                              hipStream_t stream) {
  (void)in_sizes; (void)n_in; (void)out_size; (void)ws_size;
  const float*   z1m   = (const float*)d_in[0];   // z1_mean   [B,N1,D]
  // d_in[1] z1_logvar: unused by reference loss
  const float*   z2m   = (const float*)d_in[2];   // z2_mean   [B,N2,D]
  // d_in[3] z2_logvar: unused
  const int*     midx  = (const int*)d_in[4];     // match_indices [B,N1]
  const float*   temp  = (const float*)d_in[5];   // temperature [1]
  const uint8_t* mask1 = (const uint8_t*)d_in[6]; // bool mask [B,N1]
  float*         ws    = (float*)d_ws;            // [0]=loss sum, [1]=count
  float*         out   = (float*)d_out;

  twin_loss_init<<<1, 1, 0, stream>>>(ws);
  // 16 batches x 8 row-tile groups; each block = 8 waves, each wave = one 16-row tile
  twin_loss_main<<<BB * (NN1 / (16 * WPB)), 32 * WPB, 0, stream>>>(z1m, z2m, midx, temp, mask1, ws);
  twin_loss_fini<<<1, 1, 0, stream>>>(ws, out);
}
